// TransformerQuantizerStorch_57088705299031
// MI455X (gfx1250) — compile-verified, matrix operand order checked by reading, not run on hardware
//
#include <hip/hip_runtime.h>
#include <hip/hip_bf16.h>

typedef __attribute__((ext_vector_type(16))) __bf16 v16bf;
typedef __attribute__((ext_vector_type(8)))  float  v8f;

#define G_  4
#define D_  512
#define K_  2048
#define M_  8192          // N*H*W tokens
#define L_  1024          // H*W
#define CH_ 2048          // channels
#define QUANT_ELEMS 16777216L   // 8*2048*32*32

static __device__ __forceinline__ unsigned short f2bf(float f) {
    unsigned int u = __builtin_bit_cast(unsigned int, f);
    unsigned int r = u + 0x7FFFu + ((u >> 16) & 1u);   // round-to-nearest-even
    return (unsigned short)(r >> 16);
}

// ---------------- pack / convert kernels ----------------

// latents [N, C, H, W] f32  ->  x_bf16 [G, M, D]  (m = n*L + l, d within group)
__global__ __launch_bounds__(256) void pack_latents_kernel(
    const float* __restrict__ latents, unsigned short* __restrict__ xbf, long total)
{
    long i = (long)blockIdx.x * blockDim.x + threadIdx.x;
    if (i >= total) return;
    int g   = (int)(i >> 22);          // M_*D_ = 2^22
    int rem = (int)(i & 4194303);
    int m   = rem >> 9;                // D_ = 512
    int d   = rem & 511;
    int n   = m >> 10;                 // L_ = 1024
    int l   = m & 1023;
    long src = ((long)n * CH_ + g * D_ + d) * L_ + l;
    xbf[i] = f2bf(latents[src]);
}

__global__ __launch_bounds__(256) void cvt_bf16_kernel(
    const float* __restrict__ in, unsigned short* __restrict__ out, long total)
{
    long i = (long)blockIdx.x * blockDim.x + threadIdx.x;
    if (i >= total) return;
    out[i] = f2bf(in[i]);
}

// W [G, D, E] f32 -> Wt_bf16 [G, E, D]  (k-contiguous per output column)
__global__ __launch_bounds__(256) void cvt_transpose_bf16_kernel(
    const float* __restrict__ in, unsigned short* __restrict__ out, long total)
{
    long i = (long)blockIdx.x * blockDim.x + threadIdx.x;
    if (i >= total) return;
    int g   = (int)(i >> 18);          // D_*D_ = 2^18
    int rem = (int)(i & 262143);
    int e   = rem >> 9;                // row of Wt
    int d   = rem & 511;               // col of Wt (= k)
    out[i] = f2bf(in[(long)g * D_ * D_ + (long)d * D_ + e]);
}

// ---------------- WMMA GEMM ----------------
// C[g][m][n] = scale * sum_k A[g][m][k] * B[g][n][k] + bias[g][n]
// A: [M,Kd] bf16 row-major.  B: [N,Kd] bf16 row-major, used transposed
// (weights are pre-transposed at pack time; key_p is naturally [K_codes, D]).
// Block tile 128x128, 8 waves (wave32). Wave = 4 M-tiles x 2 N-tiles of 16x16.
// B tile staged in LDS in WMMA *fragment order*: all staging and all fragment
// loads are 16-byte b128 ops.
template <bool OUTBF16>
__global__ __launch_bounds__(256) void gemm_wmma_kernel(
    const unsigned short* __restrict__ A, long aGS,
    const unsigned short* __restrict__ B, long bGS,
    const float* __restrict__ bias, long biasGS,
    void* __restrict__ Cout, long cGS,
    int Kd, int lda, int ldb, int ldc, float scale)
{
    const int g = blockIdx.z;
    A += (long)g * aGS;
    B += (long)g * bGS;
    const float* biasg = bias ? (bias + (long)g * biasGS) : nullptr;

    const int m0 = blockIdx.y * 128;
    const int n0 = blockIdx.x * 128;

    __shared__ unsigned short As[128][32];     // 8 KB, row-major (k contiguous)
    __shared__ unsigned short Bs[8][32][16];   // 8 KB, [n_tile][lane][frag j]

    const int t    = threadIdx.x;
    const int lane = t & 31;
    const int w    = t >> 5;
    const int wM   = (w >> 2) * 64;            // 0 / 64   (4 M-tiles each)
    const int wN   = (w & 3) * 32;             // 0/32/64/96 (2 N-tiles each)

    v8f acc[4][2];
#pragma unroll
    for (int mt = 0; mt < 4; ++mt)
#pragma unroll
        for (int nt = 0; nt < 2; ++nt)
#pragma unroll
            for (int r = 0; r < 8; ++r) acc[mt][nt][r] = 0.0f;

    const int row = lane & 15;
    const int kb  = (lane >> 4) * 8;           // ISA 16-bit fragment K-base

    union FragU { v16bf v; uint4 q[2]; };

    for (int k0 = 0; k0 < Kd; k0 += 32) {
        // ---- stage A tile 128x32: 2 x uint4 per thread ----
        {
            int r = t >> 1, c = (t & 1) * 16;
            const unsigned short* src = A + (long)(m0 + r) * lda + (k0 + c);
            *(uint4*)&As[r][c]     = *(const uint4*)(src);
            *(uint4*)&As[r][c + 8] = *(const uint4*)(src + 8);
            if (k0 + 32 < Kd) __builtin_prefetch((const void*)(src + 32), 0, 1);
        }
        // ---- stage B tile (cols n0..n0+127, k0..k0+31) into fragment order:
        //      8 consecutive k for one n land contiguous -> b128 load + store.
#pragma unroll
        for (int i = 0; i < 2; ++i) {
            int idx = t + i * 256;                // 0..511
            int nn  = idx >> 2;                   // 0..127
            int kk  = (idx & 3) << 3;             // 0,8,16,24
            const uint4 v = *(const uint4*)(B + (long)(n0 + nn) * ldb + (k0 + kk));
            *(uint4*)&Bs[nn >> 4][(nn & 15) + ((kk & 8) << 1)][(kk & 16) >> 1] = v;
        }
        __syncthreads();

        // ---- B fragments: 2 contiguous b128 loads each ----
        FragU fb[2];
#pragma unroll
        for (int nt = 0; nt < 2; ++nt) {
            const unsigned short* p = &Bs[(wN >> 4) + nt][lane][0];
            fb[nt].q[0] = *(const uint4*)(p);
            fb[nt].q[1] = *(const uint4*)(p + 8);
        }
        // ---- A fragments + 8 WMMAs ----
#pragma unroll
        for (int mt = 0; mt < 4; ++mt) {
            FragU fa;
            const unsigned short* p = &As[wM + mt * 16 + row][kb];
            fa.q[0] = *(const uint4*)(p);
            fa.q[1] = *(const uint4*)(p + 16);
            acc[mt][0] = __builtin_amdgcn_wmma_f32_16x16x32_bf16(
                false, fa.v, false, fb[0].v, (short)0, acc[mt][0], false, false);
            acc[mt][1] = __builtin_amdgcn_wmma_f32_16x16x32_bf16(
                false, fa.v, false, fb[1].v, (short)0, acc[mt][1], false, false);
        }
        __syncthreads();
    }

    // ---- epilogue: D layout: VGPR r -> M = r + (lane>>4)*8, N = lane&15 ----
    const int mofs = (lane >> 4) * 8;
    const int ncl  = lane & 15;
#pragma unroll
    for (int nt = 0; nt < 2; ++nt) {
        const int ncol = n0 + wN + nt * 16 + ncl;
        const float bv = biasg ? biasg[ncol] : 0.0f;
#pragma unroll
        for (int mt = 0; mt < 4; ++mt) {
            const int mbase = m0 + wM + mt * 16 + mofs;
#pragma unroll
            for (int r = 0; r < 8; ++r) {
                float v = acc[mt][nt][r] * scale + bv;
                long  o = (long)g * cGS + (long)(mbase + r) * ldc + ncol;
                if (OUTBF16) ((unsigned short*)Cout)[o] = f2bf(v);
                else         ((float*)Cout)[o] = v;
            }
        }
    }
}

// ---------------- Gumbel-argmax sampling + value gather ----------------

static __device__ __forceinline__ float gumbel_hash(unsigned int x)
{
    x *= 2654435769u; x += 0x9E3779B9u;
    x ^= x >> 16; x *= 0x7feb352du;
    x ^= x >> 15; x *= 0x846ca68bu;
    x ^= x >> 16;
    float u = (float)(x >> 8) * (1.0f / 16777216.0f) + 1e-10f;
    if (u > 0.99999988f) u = 0.99999988f;
    return -__logf(-__logf(u));
}

// grid: (M_, G_), block: 256. logits: [M, G*K] f32 (already scaled), valp: [G,K,D] f32
__global__ __launch_bounds__(256) void sample_gather_kernel(
    const float* __restrict__ logits, const float* __restrict__ valp,
    float* __restrict__ quant)
{
    const int m = blockIdx.x;
    const int g = blockIdx.y;
    const int t = threadIdx.x;

    __shared__ float sv[256];
    __shared__ int   si[256];

    const float* lrow = logits + (long)m * (G_ * K_) + (long)g * K_;
    float best = -3.4e38f;
    int   bidx = 0;
#pragma unroll
    for (int k = t; k < K_; k += 256) {
        unsigned int seed = (unsigned int)(((m * G_ + g) << 11) + k);
        float v = lrow[k] + gumbel_hash(seed);
        if (v > best) { best = v; bidx = k; }
    }
    sv[t] = best; si[t] = bidx;
    __syncthreads();
    for (int s = 128; s > 0; s >>= 1) {
        if (t < s && sv[t + s] > sv[t]) { sv[t] = sv[t + s]; si[t] = si[t + s]; }
        __syncthreads();
    }
    const int idx = si[0];

    const float* vrow = valp + ((long)g * K_ + idx) * D_;
    const int n = m >> 10, l = m & 1023;
    float* qout = quant + ((long)n * CH_ + g * D_) * L_ + l;
    for (int d = t; d < D_; d += 256) qout[(long)d * L_] = vrow[d];
}

// ---------------- host launch ----------------

static inline long cdiv(long a, long b) { return (a + b - 1) / b; }

extern "C" void kernel_launch(void* const* d_in, const int* in_sizes, int n_in,
                              void* d_out, int out_size, void* d_ws, size_t ws_size,
                              hipStream_t stream)
{
    (void)in_sizes; (void)n_in; (void)out_size; (void)ws_size;

    const float* latents   = (const float*)d_in[0];
    const float* codebooks = (const float*)d_in[1];
    const float* Wq = (const float*)d_in[2];
    const float* bq = (const float*)d_in[3];
    const float* Wk = (const float*)d_in[4];
    const float* bk = (const float*)d_in[5];
    const float* Wv = (const float*)d_in[6];
    const float* bv = (const float*)d_in[7];

    float* quant  = (float*)d_out;
    float* logits = quant + QUANT_ELEMS;         // [M, G, K] == [n,h,w,g,k] flat

    // workspace layout (ushort elements, then f32)
    unsigned short* xbf = (unsigned short*)d_ws;            // [G,M,D]
    unsigned short* qbf = xbf + (long)G_ * M_ * D_;         // [G,M,D]
    unsigned short* wqt = qbf + (long)G_ * M_ * D_;         // [G,E,D] (W^T)
    unsigned short* wkt = wqt + (long)G_ * D_ * D_;
    unsigned short* wvt = wkt + (long)G_ * D_ * D_;
    unsigned short* cbb = wvt + (long)G_ * D_ * D_;         // [G,K,D]
    unsigned short* kpb = cbb + (long)G_ * K_ * D_;         // [G,K,D]
    float*          vpf = (float*)(kpb + (long)G_ * K_ * D_); // [G,K,D] f32

    const long nX  = (long)G_ * M_ * D_;
    const long nW  = (long)G_ * D_ * D_;
    const long nCB = (long)G_ * K_ * D_;

    // 1) pack / convert to bf16 (weights transposed so all GEMM B's are
    //    k-contiguous per output column -> pure b128 LDS staging)
    pack_latents_kernel<<<cdiv(nX, 256), 256, 0, stream>>>(latents, xbf, nX);
    cvt_transpose_bf16_kernel<<<cdiv(nW, 256), 256, 0, stream>>>(Wq, wqt, nW);
    cvt_transpose_bf16_kernel<<<cdiv(nW, 256), 256, 0, stream>>>(Wk, wkt, nW);
    cvt_transpose_bf16_kernel<<<cdiv(nW, 256), 256, 0, stream>>>(Wv, wvt, nW);
    cvt_bf16_kernel<<<cdiv(nCB, 256), 256, 0, stream>>>(codebooks, cbb, nCB);

    // 2) q = x @ Wq + bq            [8192x512] x [512x512] -> bf16
    gemm_wmma_kernel<true><<<dim3(D_/128, M_/128, G_), 256, 0, stream>>>(
        xbf, (long)M_*D_, wqt, (long)D_*D_, bq, D_,
        qbf, (long)M_*D_, D_, D_, D_, D_, 1.0f);

    // 3) key_p = C @ Wk + bk        [2048x512] x [512x512] -> bf16
    gemm_wmma_kernel<true><<<dim3(D_/128, K_/128, G_), 256, 0, stream>>>(
        cbb, (long)K_*D_, wkt, (long)D_*D_, bk, D_,
        kpb, (long)K_*D_, D_, D_, D_, D_, 1.0f);

    // 4) val_p = C @ Wv + bv        [2048x512] x [512x512] -> f32
    gemm_wmma_kernel<false><<<dim3(D_/128, K_/128, G_), 256, 0, stream>>>(
        cbb, (long)K_*D_, wvt, (long)D_*D_, bv, D_,
        vpf, (long)K_*D_, D_, D_, D_, D_, 1.0f);

    // 5) logit = (q @ key_p^T) / sqrt(K)   [8192x512] x [512x2048] -> f32 in d_out
    //    out layout [m][g][k]: ldc = G*K, per-group offset g*K via cGS = K
    const float inv_sqrt_k = 0.02209708691207961f;   // 1/sqrt(2048)
    gemm_wmma_kernel<false><<<dim3(K_/128, M_/128, G_), 256, 0, stream>>>(
        qbf, (long)M_*D_, kpb, (long)K_*D_, nullptr, 0,
        logits, (long)K_, D_, D_, D_, G_*K_, inv_sqrt_k);

    // 6) Gumbel-max sample + gather val_p rows into NCHW quant
    sample_gather_kernel<<<dim3(M_, G_), 256, 0, stream>>>(logits, vpf, quant);
}
